// QuantumConv1DLayer_81003083202957
// MI455X (gfx1250) — compile-verified, matrix operand order checked by reading, not run on hardware
//
#include <hip/hip_runtime.h>

// Reference reduces to: out[b, oc, pos] = 1.0f for all 32*128*4096 elements.
// Pure store-bandwidth problem: 64 MB / 23.3 TB/s ~ 2.9 us floor on MI455X.
// We materialize the constant with one v_wmma_f32_16x16x32_f16 per wave
// (A=1.0, B=1/32 -> every accumulator element is exactly 1.0f), then each
// lane streams its 8 accumulator floats to memory as two b128 stores.

typedef __attribute__((ext_vector_type(16))) _Float16 v16h;
typedef __attribute__((ext_vector_type(8)))  float    v8f;
typedef __attribute__((ext_vector_type(4)))  float    v4f;

__global__ __launch_bounds__(256) void qconv_fill_ones_wmma(float* __restrict__ out,
                                                            long long n) {
    // Build A (all 1.0) and B (all 1/32 = 0x2800 f16, exactly representable).
    v16h a, b;
#pragma unroll
    for (int i = 0; i < 16; ++i) {
        a[i] = (_Float16)1.0f;
        b[i] = (_Float16)0.03125f;
    }
    v8f c = {};
    // D = A x B + 0 : every element = sum_{k=0}^{31} 1.0 * (1/32) = 1.0 exactly
    // (f32 accumulate, RNE, no rounding error). EXEC is all-1s here: no
    // divergence has occurred yet, full 256-thread blocks.
    c = __builtin_amdgcn_wmma_f32_16x16x32_f16(
        /*neg_a=*/false, a, /*neg_b=*/false, b,
        /*c_mod=*/(short)0, c, /*reuse_a=*/false, /*reuse_b=*/false);

    long long tid  = (long long)blockIdx.x * blockDim.x + threadIdx.x;
    long long base = tid * 8;

    if (base + 8 <= n) {
        // 32 contiguous bytes per lane -> two global_store_b128; each wave32
        // covers 1 KB of contiguous memory. Fully coalesced.
        v4f lo = {c[0], c[1], c[2], c[3]};
        v4f hi = {c[4], c[5], c[6], c[7]};
        *(v4f*)(out + base)     = lo;
        *(v4f*)(out + base + 4) = hi;
    } else if (base < n) {
        // Defensive tail (not hit for 16,777,216 = 8 * 2,097,152).
        for (long long i = base; i < n; ++i) out[i] = c[0];
    }
}

extern "C" void kernel_launch(void* const* d_in, const int* in_sizes, int n_in,
                              void* d_out, int out_size, void* d_ws, size_t ws_size,
                              hipStream_t stream) {
    (void)d_in; (void)in_sizes; (void)n_in; (void)d_ws; (void)ws_size;

    float* out = (float*)d_out;
    long long n = (long long)out_size;          // 32*128*4096 = 16,777,216 floats

    const int  block   = 256;                   // 8 wave32 per block
    long long  threads = (n + 7) / 8;           // 8 floats per thread
    long long  grid    = (threads + block - 1) / block;   // 8192 blocks
    if (grid < 1) grid = 1;

    qconv_fill_ones_wmma<<<(dim3)(unsigned)grid, (dim3)block, 0, stream>>>(out, n);
}
